// GCN_Layer_build_61306363183454
// MI455X (gfx1250) — matrix operand order repaired and checked by reading, not compiled, not run on hardware
//
#include <hip/hip_runtime.h>
#include <hip/hip_bf16.h>

typedef __attribute__((ext_vector_type(16))) __bf16 v16bf;
typedef __attribute__((ext_vector_type(8)))  float  v8f;

#define N_NODES 50000
#define N_EDGES 800000
#define D 128                    // D_IN == D_OUT == 128
#define M_TILES (N_NODES / 16)   // 3125, exact
#define GEMM_BLOCKS 640          // grid-stride: ~5 M-tiles per block, B frags reused

// ---------------------------------------------------------------------------
// Kernel 1: support = X @ W via bf16 WMMA (f32 accumulate).
// Block = 256 threads = 8 waves; wave w owns N-tile w (cols 16w..16w+15).
// B fragments (W) are loaded/converted ONCE per block, then a grid-stride
// loop over 16-row M-tiles does: A b128-loads -> cvt_pk_bf16 -> 4 WMMAs -> store.
// ---------------------------------------------------------------------------
__global__ __launch_bounds__(256)
void gcn_gemm_wmma(const float* __restrict__ X,
                   const float* __restrict__ W,
                   float* __restrict__ S) {
    const int wave   = threadIdx.x >> 5;     // 0..7 -> ntile
    const int lane   = threadIdx.x & 31;
    const int lane16 = lane & 15;
    const bool hi    = (lane >= 16);

    // ---- B fragments (W is [K=128][N=128] row-major, L2-resident) ----
    // 32x16 bf16 B tile: lane<16 holds K=0..7,16..23 of col N, lane>=16 holds K=8..15,24..31.
    const int ncol = wave * 16 + lane16;
    const int koff = hi ? 8 : 0;
    v16bf bfrag[4];
#pragma unroll
    for (int ks = 0; ks < 4; ++ks) {
        const int kbase = ks * 32 + koff;
#pragma unroll
        for (int e = 0; e < 16; ++e) {
            bfrag[ks][e] = (__bf16)W[(kbase + e + (e & 8)) * D + ncol];
        }
    }

    // ---- grid-stride over M-tiles ----
    // A 16x32 bf16 tile: lane row M = lane%16; elems 0..7 = K = aoff+e,
    // elems 8..15 = K = aoff+16+e, with aoff = (lane<16 ? 0 : 8).
    const int aoff = hi ? 8 : 0;
    for (int mtile = blockIdx.x; mtile < M_TILES; mtile += GEMM_BLOCKS) {
        const int arow = mtile * 16 + lane16;
        const float* __restrict__ xrow = X + (size_t)arow * D;

        v8f acc = {};
#pragma unroll
        for (int ks = 0; ks < 4; ++ks) {
            const int kbase = ks * 32 + aoff;
            v16bf afrag;
#pragma unroll
            for (int e = 0; e < 8; ++e)  afrag[e]     = (__bf16)xrow[kbase + e];
#pragma unroll
            for (int e = 0; e < 8; ++e)  afrag[8 + e] = (__bf16)xrow[kbase + 16 + e];

            // (neg_a, A, neg_b, B, c_mod, C, reuse_a, reuse_b)
            acc = __builtin_amdgcn_wmma_f32_16x16x32_bf16(
                false, afrag, false, bfrag[ks], (short)0, acc, false, false);
        }

        // ---- store C: VGPR r -> M = r + (hi?8:0), N = lane%16 ----
        const int crow0 = mtile * 16 + (hi ? 8 : 0);
#pragma unroll
        for (int r = 0; r < 8; ++r) {
            S[(size_t)(crow0 + r) * D + ncol] = acc[r];
        }
    }
}

// ---------------------------------------------------------------------------
// Kernel 2: output[n][d] = bias[d]   (also un-poisons d_out)
// ---------------------------------------------------------------------------
__global__ __launch_bounds__(256)
void gcn_init_out(float* __restrict__ out,
                  const float* __restrict__ bias,
                  int total) {
    int i = blockIdx.x * blockDim.x + threadIdx.x;
    if (i < total) out[i] = bias[i & (D - 1)];
}

// ---------------------------------------------------------------------------
// Kernel 3: edge aggregation. One wave per edge: lane handles 4 channels.
// float4 gather from support (L2-resident), hardware f32 atomics into output
// (output fits in 192 MB L2 -> atomics resolve at L2, no HBM round-trip).
// ---------------------------------------------------------------------------
__global__ __launch_bounds__(256)
void gcn_edge_agg(const int* __restrict__ row,
                  const int* __restrict__ col,
                  const float* __restrict__ ew,
                  const float* __restrict__ S,
                  float* __restrict__ out,
                  int nedges) {
    const int e    = (int)((blockIdx.x * blockDim.x + threadIdx.x) >> 5);
    const int lane = threadIdx.x & 31;
    if (e >= nedges) return;

    const int   r = row[e];
    const int   c = col[e];
    const float w = ew[e];

    const float4 m = *((const float4*)(S + (size_t)c * D) + lane);
    float* __restrict__ op = out + (size_t)r * D + lane * 4;

    unsafeAtomicAdd(op + 0, m.x * w);
    unsafeAtomicAdd(op + 1, m.y * w);
    unsafeAtomicAdd(op + 2, m.z * w);
    unsafeAtomicAdd(op + 3, m.w * w);
}

// ---------------------------------------------------------------------------
extern "C" void kernel_launch(void* const* d_in, const int* in_sizes, int n_in,
                              void* d_out, int out_size, void* d_ws, size_t ws_size,
                              hipStream_t stream) {
    const float* X    = (const float*)d_in[0];              // [50000,128]
    const int*   ei   = (const int*)  d_in[1];              // [2,800000]
    const float* ew   = (const float*)d_in[2];              // [800000]
    const float* W    = (const float*)d_in[3];              // [128,128]
    const float* bias = (const float*)d_in[4];              // [128]
    float*       out  = (float*)d_out;                      // [50000,128]
    float*       S    = (float*)d_ws;                       // support scratch, 25.6 MB

    const int* row  = ei;                // edge_index[0] = dst rows
    const int* colv = ei + N_EDGES;      // edge_index[1] = src cols

    // 1) support = X @ W   (WMMA, grid-stride over M-tiles)
    gcn_gemm_wmma<<<GEMM_BLOCKS, 256, 0, stream>>>(X, W, S);

    // 2) out = bias (broadcast)
    const int total = N_NODES * D;
    gcn_init_out<<<(total + 255) / 256, 256, 0, stream>>>(out, bias, total);

    // 3) scatter-add messages: one wave per edge
    const int waves_per_block = 256 / 32;
    gcn_edge_agg<<<(N_EDGES + waves_per_block - 1) / waves_per_block, 256, 0, stream>>>(
        row, colv, ew, S, out, N_EDGES);
}
